// DCRNN_3994319585701
// MI455X (gfx1250) — matrix operand order repaired and checked by reading, not compiled, not run on hardware
//
#include <hip/hip_runtime.h>
#include <hip/hip_bf16.h>

// ---------------------------------------------------------------------------
// DCRNN on MI455X (gfx1250): diffusion einsum batched to 1024x1024x(B*d) GEMMs
// executed with v_wmma_f32_16x16x32_f16 (f16 in, f32 accum).
// A-tiles staged with global_load_async_to_lds_b128 (ASYNCcnt path).
// ---------------------------------------------------------------------------

#define NB_    16      // batch
#define NT_    12      // encoder timesteps
#define NN_    1024    // graph nodes
#define NH_    64      // hidden
#define NPRED_ 12

typedef __attribute__((ext_vector_type(16))) _Float16 v16h;
typedef __attribute__((ext_vector_type(8)))  _Float16 v8h;
typedef __attribute__((ext_vector_type(8)))  float    v8f;

// ---------------------------------------------------------------------------
// Generic f16 WMMA GEMM:  C(MxNc) = A(MxK) @ B(KxNc), row-major, K%32==0,
// M%128==0.  Workgroup 256 thr = 8 waves; block tile 128x128; wave tile 32x64.
// mode 0: f32 out (+bias, +activation)      [weight matmul]
// mode 1: f16 out to Xtmp (pitch Nc) and scatter into Hcat concat buffer
//         at feature-column offset colOff   [diffusion hop]
// ---------------------------------------------------------------------------
__global__ __launch_bounds__(256) void gemm_f16_wmma(
    const _Float16* __restrict__ A, int lda,
    const _Float16* __restrict__ Bm, int ldb,
    int M, int Nc, int K,
    int mode, float* __restrict__ Cout, int ldc,
    const float* __restrict__ bias, int act,
    _Float16* __restrict__ Xtmp,
    _Float16* __restrict__ Hcat, int Kp, int colOff, int dDim)
{
    __shared__ __align__(16) _Float16 lA [128 * 40]; // 128 rows x 32 K, pitch 40
    __shared__ __align__(16) _Float16 lBt[128 * 40]; // 128 cols x 32 K, pitch 40

    const int tid   = threadIdx.x;
    const int lane  = tid & 31;
    const int w     = tid >> 5;
    const int wm    = w & 3;        // 4 waves along M
    const int wn    = w >> 2;       // 2 waves along N
    const int gm0   = blockIdx.y * 128;
    const int gn0   = blockIdx.x * 128;
    const int mlane = lane & 15;
    const int half  = lane >> 4;

    v8f acc[2][4];
#pragma unroll
    for (int i = 0; i < 2; ++i)
#pragma unroll
        for (int j = 0; j < 4; ++j)
#pragma unroll
            for (int v = 0; v < 8; ++v) acc[i][j][v] = 0.0f;

    for (int kb = 0; kb < K; kb += 32) {
        // ---- stage A tile 128x32 via async global->LDS (no VGPR bounce) ----
        // 512 chunks of 16B; 2 per thread.
        {
#pragma unroll
            for (int r = 0; r < 2; ++r) {
                const int task = tid + r * 256;
                const int row  = task >> 2;
                const int ch   = task & 3;           // 4 x 8-f16 chunks per row
                const _Float16* gp =
                    A + (size_t)(gm0 + row) * lda + kb + ch * 8;
                const unsigned loff =
                    (unsigned)(uintptr_t)&lA[row * 40 + ch * 8];
                asm volatile("global_load_async_to_lds_b128 %0, %1, off"
                             :: "v"(loff), "v"(gp) : "memory");
            }
        }
        // ---- stage B tile 32x128 transposed: coalesced read, write [n][k] --
        {
#pragma unroll
            for (int r = 0; r < 2; ++r) {
                const int task = tid + r * 256;
                const int k    = task >> 4;          // 0..31
                const int nch  = task & 15;          // 16 chunks of 8 cols
                const _Float16* gp =
                    Bm + (size_t)(kb + k) * ldb + gn0 + nch * 8;
                v8h bv = *(const v8h*)gp;
                if (kb + 32 < K && r == 0)
                    __builtin_prefetch(gp + (size_t)32 * ldb, 0, 1);
#pragma unroll
                for (int j = 0; j < 8; ++j)
                    lBt[(nch * 8 + j) * 40 + k] = bv[j];
            }
        }
        asm volatile("s_wait_asynccnt 0" ::: "memory");
        __syncthreads();

        // ---- build fragments per ISA 16-bit A/B lane layouts ---------------
        v16h af[2], bf[4];
#pragma unroll
        for (int i = 0; i < 2; ++i) {
            const int m = wm * 32 + i * 16 + mlane;
            v8h lo = *(const v8h*)&lA[m * 40 + half * 8];      // K 0..7 / 8..15
            v8h hi = *(const v8h*)&lA[m * 40 + 16 + half * 8]; // K 16..23 / 24..31
            af[i] = __builtin_shufflevector(lo, hi,
                    0,1,2,3,4,5,6,7,8,9,10,11,12,13,14,15);
        }
#pragma unroll
        for (int j = 0; j < 4; ++j) {
            const int n = wn * 64 + j * 16 + mlane;
            v8h lo = *(const v8h*)&lBt[n * 40 + half * 8];      // K 0..7 / 8..15
            v8h hi = *(const v8h*)&lBt[n * 40 + 16 + half * 8]; // K 16..23 / 24..31
            bf[j] = __builtin_shufflevector(lo, hi,
                    0,1,2,3,4,5,6,7,8,9,10,11,12,13,14,15);
        }

#pragma unroll
        for (int i = 0; i < 2; ++i)
#pragma unroll
            for (int j = 0; j < 4; ++j)
                acc[i][j] = __builtin_amdgcn_wmma_f32_16x16x32_f16(
                    false, af[i], false, bf[j], (short)0, acc[i][j],
                    false, false);
        // make sure every wave's fragment ds_loads drained before next-iter
        // async writes can land in lA/lBt
        asm volatile("s_wait_dscnt 0" ::: "memory");
        __syncthreads();
    }

    // ---- epilogue: C/D layout = VGPR v -> M=v (lanes0-15) / v+8 (lanes16-31)
    const int rowoff = half * 8;
#pragma unroll
    for (int i = 0; i < 2; ++i) {
#pragma unroll
        for (int j = 0; j < 4; ++j) {
            const int rbase = gm0 + wm * 32 + i * 16;
            const int col   = gn0 + wn * 64 + j * 16 + mlane;
            if (col >= Nc) continue;
#pragma unroll
            for (int v = 0; v < 8; ++v) {
                const int row = rbase + v + rowoff;
                float val = acc[i][j][v];
                if (mode == 0) {
                    if (bias) val += bias[col];
                    if (act == 1)      val = 1.0f / (1.0f + expf(-val));
                    else if (act == 2) val = tanhf(val);
                    Cout[(size_t)row * ldc + col] = val;
                } else {
                    const _Float16 hv = (_Float16)val;
                    if (Xtmp) Xtmp[(size_t)row * Nc + col] = hv;
                    const int b  = col / dDim;
                    const int dd = col - b * dDim;
                    Hcat[((size_t)row * NB_ + b) * Kp + colOff + dd] = hv;
                }
            }
        }
    }
}

// ---------------------------------------------------------------------------
// Build Xc = concat([inp, (r?)*state]) in (N, B*d) f16; also write feature-0
// slice of the concat buffer Hcat[(n*B+b)][0..d).
// inp addressed as inp[b*sb + off + n*sn + i].
// ---------------------------------------------------------------------------
__global__ void build_xc(const float* __restrict__ inp, int sb, int off, int sn,
                         int di, const float* __restrict__ st,
                         const float* __restrict__ rgate,
                         _Float16* __restrict__ Xc, _Float16* __restrict__ Hcat,
                         int d, int C, int Kp)
{
    const int idx = blockIdx.x * blockDim.x + threadIdx.x;
    if (idx >= NN_ * C) return;
    const int n = idx / C;
    const int c = idx - n * C;
    const int b = c / d;
    const int dd = c - b * d;
    float v;
    if (dd < di) {
        v = inp[(size_t)b * sb + off + (size_t)n * sn + dd];
    } else {
        const int h = dd - di;
        v = st[((size_t)n * NB_ + b) * NH_ + h];
        if (rgate) v *= rgate[((size_t)n * NB_ + b) * (2 * NH_) + h];
    }
    const _Float16 hv = (_Float16)v;
    Xc[(size_t)n * C + c] = hv;
    Hcat[((size_t)n * NB_ + b) * Kp + dd] = hv;
}

__global__ void zero_pad_hcat(_Float16* __restrict__ Hcat, int Kp, int K)
{
    const int padw = Kp - K;
    const int idx = blockIdx.x * blockDim.x + threadIdx.x;
    if (idx >= NN_ * NB_ * padw) return;
    const int r = idx / padw;
    const int p = idx - r * padw;
    Hcat[(size_t)r * Kp + K + p] = (_Float16)0.0f;
}

// new_state = u*state + (1-u)*c   (u = sigmoided RU[..., H:2H], c = tanh'ed)
__global__ void gru_combine(const float* __restrict__ RU,
                            const float* __restrict__ Cc,
                            float* __restrict__ st)
{
    const int idx = blockIdx.x * blockDim.x + threadIdx.x;
    if (idx >= NN_ * NB_ * NH_) return;
    const int r = idx >> 6;
    const int h = idx & 63;
    const float u = RU[(size_t)r * (2 * NH_) + NH_ + h];
    st[idx] = u * st[idx] + (1.0f - u) * Cc[idx];
}

// out[b, t, n, 0] = st1[n,b,:] @ W + b0 ; also feed next decoder input
__global__ void proj_out(const float* __restrict__ st1,
                         const float* __restrict__ W,
                         const float* __restrict__ b0,
                         float* __restrict__ out, float* __restrict__ dec_in,
                         int t)
{
    const int idx = blockIdx.x * blockDim.x + threadIdx.x;
    if (idx >= NN_ * NB_) return;
    const int n = idx >> 4;
    const int b = idx & 15;
    float s = b0[0];
    const float* p = st1 + (size_t)idx * NH_;
#pragma unroll
    for (int h = 0; h < NH_; ++h) s += p[h] * W[h];
    out[((size_t)b * NPRED_ + t) * NN_ + n] = s;
    dec_in[idx] = s;
}

__global__ void f32_to_f16_k(const float* __restrict__ in,
                             _Float16* __restrict__ out, int n)
{
    const int i = blockIdx.x * blockDim.x + threadIdx.x;
    if (i < n) out[i] = (_Float16)in[i];
}

__global__ void pack_w(const float* __restrict__ W, _Float16* __restrict__ Wp,
                       int K, int Kp, int dout)
{
    const int i = blockIdx.x * blockDim.x + threadIdx.x;
    if (i >= Kp * dout) return;
    const int k = i / dout;
    Wp[i] = (k < K) ? (_Float16)W[i] : (_Float16)0.0f;
}

__global__ void zero_f32(float* __restrict__ p, int n)
{
    const int i = blockIdx.x * blockDim.x + threadIdx.x;
    if (i < n) p[i] = 0.0f;
}

// ---------------------------------------------------------------------------
extern "C" void kernel_launch(void* const* d_in, const int* in_sizes, int n_in,
                              void* d_out, int out_size, void* d_ws,
                              size_t ws_size, hipStream_t stream)
{
    (void)in_sizes; (void)n_in; (void)out_size; (void)ws_size;
    const float* inputs   = (const float*)d_in[0];
    const float* supports = (const float*)d_in[1];
    const float* outW     = (const float*)d_in[18];
    const float* outB     = (const float*)d_in[19];
    float* out = (float*)d_out;

    struct Layer {
        const float *ruW, *rub, *cW, *cb;
        int di, d, K5, Kp;
        _Float16 *WpRu, *WpC;
    };
    Layer L[4];
    for (int l = 0; l < 4; ++l) {
        L[l].ruW = (const float*)d_in[2 + 4 * l + 0];
        L[l].rub = (const float*)d_in[2 + 4 * l + 1];
        L[l].cW  = (const float*)d_in[2 + 4 * l + 2];
        L[l].cb  = (const float*)d_in[2 + 4 * l + 3];
    }
    L[0].di = 2;  L[1].di = NH_;  L[2].di = 1;  L[3].di = NH_;
    for (int l = 0; l < 4; ++l) {
        L[l].d  = L[l].di + NH_;
        L[l].K5 = 5 * L[l].d;
        L[l].Kp = ((L[l].K5 + 31) / 32) * 32;
    }

    // ---- carve workspace ---------------------------------------------------
    char* wsp = (char*)d_ws;
    auto carve = [&](size_t bytes) -> void* {
        void* p = (void*)wsp;
        wsp += (bytes + 255) & ~(size_t)255;
        return p;
    };
    _Float16* S16  = (_Float16*)carve((size_t)2 * NN_ * NN_ * 2);   // 4 MB
    for (int l = 0; l < 4; ++l) {
        L[l].WpRu = (_Float16*)carve((size_t)L[l].Kp * 2 * NH_ * 2);
        L[l].WpC  = (_Float16*)carve((size_t)L[l].Kp * NH_ * 2);
    }
    const int Cmax = NB_ * 128;                                     // 2048
    _Float16* Xc   = (_Float16*)carve((size_t)NN_ * Cmax * 2);      // 4 MB
    _Float16* Xt   = (_Float16*)carve((size_t)NN_ * Cmax * 2);      // 4 MB
    _Float16* Hcat = (_Float16*)carve((size_t)NN_ * NB_ * 640 * 2); // 21 MB
    float* RU   = (float*)carve((size_t)NN_ * NB_ * 2 * NH_ * 4);   // 8 MB
    float* Cc   = (float*)carve((size_t)NN_ * NB_ * NH_ * 4);       // 4 MB
    float* st0  = (float*)carve((size_t)NN_ * NB_ * NH_ * 4);
    float* st1  = (float*)carve((size_t)NN_ * NB_ * NH_ * 4);
    float* dcin = (float*)carve((size_t)NN_ * NB_ * 4);

    auto nb = [](int n) { return (n + 255) / 256; };

    auto gemm = [&](const _Float16* A, int lda, const _Float16* Bm, int ldb,
                    int M, int Nc, int K, int mode, float* Co, int ldc,
                    const float* bias, int act, _Float16* Xo, _Float16* Hc,
                    int Kp, int colOff, int dDim) {
        dim3 g((Nc + 127) / 128, M / 128);
        gemm_f16_wmma<<<g, 256, 0, stream>>>(A, lda, Bm, ldb, M, Nc, K, mode,
                                             Co, ldc, bias, act, Xo, Hc, Kp,
                                             colOff, dDim);
    };

    // ---- one-time (per launch) setup --------------------------------------
    f32_to_f16_k<<<nb(2 * NN_ * NN_), 256, 0, stream>>>(supports, S16,
                                                        2 * NN_ * NN_);
    for (int l = 0; l < 4; ++l) {
        pack_w<<<nb(L[l].Kp * 2 * NH_), 256, 0, stream>>>(
            L[l].ruW, L[l].WpRu, L[l].K5, L[l].Kp, 2 * NH_);
        pack_w<<<nb(L[l].Kp * NH_), 256, 0, stream>>>(
            L[l].cW, L[l].WpC, L[l].K5, L[l].Kp, NH_);
    }
    zero_f32<<<nb(NN_ * NB_ * NH_), 256, 0, stream>>>(st0, NN_ * NB_ * NH_);
    zero_f32<<<nb(NN_ * NB_ * NH_), 256, 0, stream>>>(st1, NN_ * NB_ * NH_);

    const _Float16* S0 = S16;
    const _Float16* S1 = S16 + (size_t)NN_ * NN_;
    const int M5 = NN_ * NB_;   // 16384 rows of the concat buffer

    // ---- one DCGRU cell ----------------------------------------------------
    auto run_cell = [&](const float* inp, int sb, int off, int sn,
                        const Layer& Lr, float* st) {
        const int d = Lr.d, C = NB_ * d, Kp = Lr.Kp, K5 = Lr.K5, di = Lr.di;
        for (int pass = 0; pass < 2; ++pass) {
            const float* rg = (pass == 0) ? nullptr : RU;
            build_xc<<<nb(NN_ * C), 256, 0, stream>>>(
                inp, sb, off, sn, di, st, rg, Xc, Hcat, d, C, Kp);
            // diffusion hops: x1=S0@Xc, x2=S0@x1, x3=S1@Xc, x4=S1@x3
            gemm(S0, NN_, Xc, C, NN_, C, NN_, 1, nullptr, 0, nullptr, 0,
                 Xt, Hcat, Kp, 1 * d, d);
            gemm(S0, NN_, Xt, C, NN_, C, NN_, 1, nullptr, 0, nullptr, 0,
                 nullptr, Hcat, Kp, 2 * d, d);
            gemm(S1, NN_, Xc, C, NN_, C, NN_, 1, nullptr, 0, nullptr, 0,
                 Xt, Hcat, Kp, 3 * d, d);
            gemm(S1, NN_, Xt, C, NN_, C, NN_, 1, nullptr, 0, nullptr, 0,
                 nullptr, Hcat, Kp, 4 * d, d);
            if (Kp > K5)
                zero_pad_hcat<<<nb(M5 * (Kp - K5)), 256, 0, stream>>>(Hcat, Kp,
                                                                      K5);
            if (pass == 0)
                gemm(Hcat, Kp, Lr.WpRu, 2 * NH_, M5, 2 * NH_, Kp, 0, RU,
                     2 * NH_, Lr.rub, /*sigmoid*/ 1, nullptr, nullptr, 0, 0, 1);
            else
                gemm(Hcat, Kp, Lr.WpC, NH_, M5, NH_, Kp, 0, Cc, NH_, Lr.cb,
                     /*tanh*/ 2, nullptr, nullptr, 0, 0, 1);
        }
        gru_combine<<<nb(NN_ * NB_ * NH_), 256, 0, stream>>>(RU, Cc, st);
    };

    // ---- encoder -----------------------------------------------------------
    for (int t = 0; t < NT_; ++t) {
        // layer 0: input = inputs[:, t, :, :]  (B,T,N,2)
        run_cell(inputs, NT_ * NN_ * 2, t * NN_ * 2, 2, L[0], st0);
        // layer 1: input = st0 (N,B,64)
        run_cell(st0, NH_, 0, NB_ * NH_, L[1], st1);
    }

    // ---- decoder (states continue from encoder finals) ---------------------
    zero_f32<<<nb(NN_ * NB_), 256, 0, stream>>>(dcin, NN_ * NB_);
    for (int t = 0; t < NPRED_; ++t) {
        run_cell(dcin, 1, 0, NB_, L[2], st0);          // input (N,B,1)
        run_cell(st0, NH_, 0, NB_ * NH_, L[3], st1);   // input (N,B,64)
        proj_out<<<nb(NN_ * NB_), 256, 0, stream>>>(st1, outW, outB, out, dcin,
                                                    t);
    }
}
